// PointNet2Decoder_60610578481763
// MI455X (gfx1250) — matrix operand order repaired and checked by reading, not compiled, not run on hardware
//
#include <hip/hip_runtime.h>

typedef _Float16 half8 __attribute__((ext_vector_type(8)));
typedef _Float16 v16h  __attribute__((ext_vector_type(16)));
typedef float    v8f   __attribute__((ext_vector_type(8)));

#define EPS_DIST 1e-8f
#define EPS_BN   1e-5f
#define CDIV(a,b) (((a)+(b)-1)/(b))

// ---------------------------------------------------------------- utilities
__global__ void zero_f32(float* p, int n) {
  int t = blockIdx.x * blockDim.x + threadIdx.x;
  if (t < n) p[t] = 0.f;
}

__global__ void cvt_f32_to_f16(const float* __restrict__ s, _Float16* __restrict__ d, int n) {
  int t = blockIdx.x * blockDim.x + threadIdx.x;
  if (t < n) d[t] = (_Float16)s[t];
}

// in (B,C,M) fp32 -> out (B,M,C) fp32
__global__ void transpose_cm(const float* __restrict__ in, float* __restrict__ out,
                             int C, int M, int total) {
  int t = blockIdx.x * blockDim.x + threadIdx.x;
  if (t >= total) return;
  int i = t % M; int r = t / M; int c = r % C; int b = r / C;
  out[((size_t)b * M + i) * C + c] = in[t];
}

// ------------------------------------------- 3-NN interpolation + concat
// unknown ux (B,n,3), known kx (B,m,3), skip (B,Cs,n), known feats transposed
// kft (B,m,Ck) -> outT (B,n,Cs+Ck) f16  (transposed layout feeds WMMA B-frags)
__global__ void interp_concat(const float* __restrict__ ux, const float* __restrict__ kx,
                              const float* __restrict__ skip, const float* __restrict__ kft,
                              _Float16* __restrict__ outT, int n, int m, int Cs, int Ck) {
  extern __shared__ float sk[];          // m*3 floats (<= 24 KB), staged once per block
  int b = blockIdx.y;
  const float* kb = kx + (size_t)b * m * 3;
  for (int i = threadIdx.x; i < m * 3; i += blockDim.x) sk[i] = kb[i];
  __syncthreads();

  int p = blockIdx.x * blockDim.x + threadIdx.x;
  if (p >= n) return;
  const float* u = ux + ((size_t)b * n + p) * 3;
  float x = u[0], y = u[1], z = u[2];
  float d0 = 3.4e38f, d1 = 3.4e38f, d2 = 3.4e38f;
  int   i0 = 0, i1 = 0, i2 = 0;
  for (int i = 0; i < m; i++) {
    float dx = x - sk[3 * i], dy = y - sk[3 * i + 1], dz = z - sk[3 * i + 2];
    float d = dx * dx + dy * dy + dz * dz;
    if (d < d0)      { d2 = d1; i2 = i1; d1 = d0; i1 = i0; d0 = d; i0 = i; }
    else if (d < d1) { d2 = d1; i2 = i1; d1 = d;  i1 = i; }
    else if (d < d2) { d2 = d;  i2 = i; }
  }
  float r0 = 1.f / (d0 + EPS_DIST), r1 = 1.f / (d1 + EPS_DIST), r2 = 1.f / (d2 + EPS_DIST);
  float rs = 1.f / (r0 + r1 + r2);
  float w0 = r0 * rs, w1 = r1 * rs, w2 = r2 * rs;

  _Float16* orow = outT + ((size_t)b * n + p) * (Cs + Ck);
  const float* sb = skip + (size_t)b * Cs * n + p;
  for (int c = 0; c < Cs; c++) orow[c] = (_Float16)sb[(size_t)c * n];
  const float* q0 = kft + ((size_t)b * m + i0) * Ck;
  const float* q1 = kft + ((size_t)b * m + i1) * Ck;
  const float* q2 = kft + ((size_t)b * m + i2) * Ck;
  for (int c = 0; c < Ck; c++)
    orow[Cs + c] = (_Float16)(w0 * q0[c] + w1 * q1[c] + w2 * q2[c]);
}

// ------------------------------------------------------------- WMMA GEMM
// Y[b](Cout,N) = W(Cout,Cin) * X[b](Cin,N), X stored transposed (B,N,Cin) f16.
// One 16x16 output tile per wave; fused per-channel sum / sum-of-squares for BN.
__global__ void gemm_wmma(const _Float16* __restrict__ W, const _Float16* __restrict__ Xt,
                          float* __restrict__ Y, float* __restrict__ sum,
                          float* __restrict__ sumsq, int Cout, int Cin, int N) {
  int lane = threadIdx.x;                    // 0..31, one wave per y-slice
  int tn   = blockIdx.x * blockDim.y + threadIdx.y;   // N tile
  int tm   = blockIdx.y;                     // Cout tile
  int b    = blockIdx.z;
  int hi   = (lane >> 4) & 1;                // lane group
  int l15  = lane & 15;
  int row  = tm * 16 + l15;                  // A-fragment row (M)
  int col  = tn * 16 + l15;                  // B-fragment column (N)

  // A 16x32 f16 frag: lanes 0-15 hold K 0-7 & 16-23; lanes 16-31 hold K 8-15 & 24-31
  const _Float16* wrow = W  + (size_t)row * Cin + (hi ? 8 : 0);
  // B 32x16 f16 frag: per-lane column, lanes 0-15 hold K 0-7 & 16-23; lanes 16-31 hold K 8-15 & 24-31
  const _Float16* xrow = Xt + ((size_t)b * N + col) * Cin + (hi ? 8 : 0);

  v8f acc;
  #pragma unroll
  for (int i = 0; i < 8; i++) acc[i] = 0.f;

  for (int k0 = 0; k0 < Cin; k0 += 32) {
    v16h a, bb;
    ((half8*)&a)[0]  = *(const half8*)(wrow + k0);        // K k0+kb .. +7
    ((half8*)&a)[1]  = *(const half8*)(wrow + k0 + 16);   // K k0+kb+16 .. +23
    ((half8*)&bb)[0] = *(const half8*)(xrow + k0);        // K k0+kb .. +7
    ((half8*)&bb)[1] = *(const half8*)(xrow + k0 + 16);   // K k0+kb+16 .. +23
    acc = __builtin_amdgcn_wmma_f32_16x16x32_f16(
        /*neg_a=*/false, a, /*neg_b=*/false, bb,
        /*c_mod=*/(short)0, acc, /*reuse_a=*/false, /*reuse_b=*/false);
  }

  // D layout: VGPR i -> row tm*16 + i (+8 for lanes 16-31), col = tn*16 + l15
  int rbase = tm * 16 + (hi ? 8 : 0);
  float* yb = Y + (size_t)b * Cout * N;
  #pragma unroll
  for (int i = 0; i < 8; i++) {
    int r = rbase + i;
    float v = acc[i];
    yb[(size_t)r * N + col] = v;
    float s = v, s2 = v * v;                 // reduce 16 columns within lane group
    #pragma unroll
    for (int off = 1; off < 16; off <<= 1) {
      s  += __shfl_xor(s,  off, 32);
      s2 += __shfl_xor(s2, off, 32);
    }
    if (l15 == 0) {
      atomicAdd(&sum[r],   s);
      atomicAdd(&sumsq[r], s2);
    }
  }
}

// ------------------------------------------------------- batchnorm params
__global__ void bn_params(const float* __restrict__ sum, const float* __restrict__ sumsq,
                          const float* __restrict__ gamma, const float* __restrict__ beta,
                          float* __restrict__ scale, float* __restrict__ shift,
                          int C, float invCount) {
  int c = blockIdx.x * blockDim.x + threadIdx.x;
  if (c >= C) return;
  float mean = sum[c] * invCount;
  float var  = sumsq[c] * invCount - mean * mean;
  float a    = gamma[c] * rsqrtf(var + EPS_BN);
  scale[c] = a;
  shift[c] = beta[c] - mean * a;
}

// ----------------------------------------------------- BN + ReLU variants
__global__ void bn_relu_f16t(const float* __restrict__ Y, const float* __restrict__ sc,
                             const float* __restrict__ sh, _Float16* __restrict__ outT,
                             int C, int N, int total) {
  int t = blockIdx.x * blockDim.x + threadIdx.x;
  if (t >= total) return;
  int nIdx = t % N; int r = t / N; int c = r % C; int b = r / C;
  float v = sc[c] * Y[t] + sh[c];
  v = v > 0.f ? v : 0.f;
  outT[((size_t)b * N + nIdx) * C + c] = (_Float16)v;
}

__global__ void bn_relu_f32t(const float* __restrict__ Y, const float* __restrict__ sc,
                             const float* __restrict__ sh, float* __restrict__ outT,
                             int C, int N, int total) {
  int t = blockIdx.x * blockDim.x + threadIdx.x;
  if (t >= total) return;
  int nIdx = t % N; int r = t / N; int c = r % C; int b = r / C;
  float v = sc[c] * Y[t] + sh[c];
  v = v > 0.f ? v : 0.f;
  outT[((size_t)b * N + nIdx) * C + c] = v;
}

__global__ void bn_relu_nchw(const float* __restrict__ Y, const float* __restrict__ sc,
                             const float* __restrict__ sh, float* __restrict__ out,
                             int C, int N, int total) {
  int t = blockIdx.x * blockDim.x + threadIdx.x;
  if (t >= total) return;
  int c = (t / N) % C;
  float v = sc[c] * Y[t] + sh[c];
  out[t] = v > 0.f ? v : 0.f;
}

// ----------------------------------------------------------- host helper
static void conv_bn(const _Float16* Wh, const _Float16* Xt, float* Y,
                    const float* gamma, const float* beta, float* bnbuf,
                    int B, int Cout, int Cin, int N, hipStream_t s) {
  zero_f32<<<2, 256, 0, s>>>(bnbuf, 512);
  gemm_wmma<<<dim3(N / 128, Cout / 16, B), dim3(32, 8), 0, s>>>(
      Wh, Xt, Y, bnbuf, bnbuf + 256, Cout, Cin, N);
  bn_params<<<1, 256, 0, s>>>(bnbuf, bnbuf + 256, gamma, beta,
                              bnbuf + 512, bnbuf + 768, Cout, 1.f / ((float)B * N));
}

extern "C" void kernel_launch(void* const* d_in, const int* in_sizes, int n_in,
                              void* d_out, int out_size, void* d_ws, size_t ws_size,
                              hipStream_t stream) {
  // setup_inputs dict order: xyz0,f0,xyz1,f1,xyz2,f2,xyz3,f3, then (w,g,b) x6
  const float* xyz0 = (const float*)d_in[0];  const float* f0 = (const float*)d_in[1];
  const float* xyz1 = (const float*)d_in[2];  const float* f1 = (const float*)d_in[3];
  const float* xyz2 = (const float*)d_in[4];  const float* f2 = (const float*)d_in[5];
  const float* xyz3 = (const float*)d_in[6];  const float* f3 = (const float*)d_in[7];
  const float* w00 = (const float*)d_in[8];   const float* g00 = (const float*)d_in[9];
  const float* b00 = (const float*)d_in[10];
  const float* w01 = (const float*)d_in[11];  const float* g01 = (const float*)d_in[12];
  const float* b01 = (const float*)d_in[13];
  const float* w10 = (const float*)d_in[14];  const float* g10 = (const float*)d_in[15];
  const float* b10 = (const float*)d_in[16];
  const float* w11 = (const float*)d_in[17];  const float* g11 = (const float*)d_in[18];
  const float* b11 = (const float*)d_in[19];
  const float* w20 = (const float*)d_in[20];  const float* g20 = (const float*)d_in[21];
  const float* b20 = (const float*)d_in[22];
  const float* w21 = (const float*)d_in[23];  const float* g21 = (const float*)d_in[24];
  const float* b21 = (const float*)d_in[25];

  constexpr int B = 8;
  constexpr int N0 = 8192, N1 = 2048, N2 = 512, N3 = 128;
  constexpr int C0 = 64, C1 = 128, C2 = 256, C3 = 512;

  // workspace arena (offsets fixed, transient buffers reused across modules)
  char* ws = (char*)d_ws;
  size_t off = 0;
  auto alloc = [&](size_t bytes) -> char* {
    char* p = ws + off;
    off = (off + bytes + 255) & ~(size_t)255;
    return p;
  };
  float*    f3t   = (float*)alloc((size_t)B * N3 * C3 * 4);       // f3 transposed (B,m,C)
  float*    f2nt  = (float*)alloc((size_t)B * N2 * 256 * 4);      // module-2 output (B,m,C)
  float*    f1nt  = (float*)alloc((size_t)B * N1 * 128 * 4);      // module-1 output (B,m,C)
  _Float16* wh20  = (_Float16*)alloc((size_t)256 * 768 * 2);
  _Float16* wh21  = (_Float16*)alloc((size_t)256 * 256 * 2);
  _Float16* wh10  = (_Float16*)alloc((size_t)256 * 384 * 2);
  _Float16* wh11  = (_Float16*)alloc((size_t)128 * 256 * 2);
  _Float16* wh00  = (_Float16*)alloc((size_t)128 * 192 * 2);
  _Float16* wh01  = (_Float16*)alloc((size_t)128 * 128 * 2);
  float*    bnbuf = (float*)alloc(4 * 256 * 4);                   // sum|sumsq|scale|shift
  _Float16* T1 = (_Float16*)alloc((size_t)B * N0 * (C0 + C1) * 2); // max concat / X2 f16
  float*    T2 = (float*)alloc((size_t)B * C1 * N0 * 4);           // max GEMM output f32
  (void)ws_size; (void)in_sizes; (void)n_in; (void)out_size;

  // weights -> f16
  cvt_f32_to_f16<<<CDIV(256 * 768, 256), 256, 0, stream>>>(w20, wh20, 256 * 768);
  cvt_f32_to_f16<<<CDIV(256 * 256, 256), 256, 0, stream>>>(w21, wh21, 256 * 256);
  cvt_f32_to_f16<<<CDIV(256 * 384, 256), 256, 0, stream>>>(w10, wh10, 256 * 384);
  cvt_f32_to_f16<<<CDIV(128 * 256, 256), 256, 0, stream>>>(w11, wh11, 128 * 256);
  cvt_f32_to_f16<<<CDIV(128 * 192, 256), 256, 0, stream>>>(w00, wh00, 128 * 192);
  cvt_f32_to_f16<<<CDIV(128 * 128, 256), 256, 0, stream>>>(w01, wh01, 128 * 128);
  // f3 (B,C3,N3) -> f3t (B,N3,C3)
  transpose_cm<<<CDIV(B * C3 * N3, 256), 256, 0, stream>>>(f3, f3t, C3, N3, B * C3 * N3);

  float* bnScale = bnbuf + 512;
  float* bnShift = bnbuf + 768;

  // ---- module 2: (xyz2, xyz3, f2, f3t) -> f2nt (B,512,256)
  interp_concat<<<dim3(N2 / 256, B), 256, N3 * 3 * 4, stream>>>(
      xyz2, xyz3, f2, f3t, T1, N2, N3, C2, C3);
  conv_bn(wh20, T1, T2, g20, b20, bnbuf, B, 256, C2 + C3, N2, stream);
  bn_relu_f16t<<<CDIV(B * 256 * N2, 256), 256, 0, stream>>>(
      T2, bnScale, bnShift, T1, 256, N2, B * 256 * N2);
  conv_bn(wh21, T1, T2, g21, b21, bnbuf, B, 256, 256, N2, stream);
  bn_relu_f32t<<<CDIV(B * 256 * N2, 256), 256, 0, stream>>>(
      T2, bnScale, bnShift, f2nt, 256, N2, B * 256 * N2);

  // ---- module 1: (xyz1, xyz2, f1, f2nt) -> f1nt (B,2048,128)
  interp_concat<<<dim3(N1 / 256, B), 256, N2 * 3 * 4, stream>>>(
      xyz1, xyz2, f1, f2nt, T1, N1, N2, C1, 256);
  conv_bn(wh10, T1, T2, g10, b10, bnbuf, B, 256, C1 + 256, N1, stream);
  bn_relu_f16t<<<CDIV(B * 256 * N1, 256), 256, 0, stream>>>(
      T2, bnScale, bnShift, T1, 256, N1, B * 256 * N1);
  conv_bn(wh11, T1, T2, g11, b11, bnbuf, B, 128, 256, N1, stream);
  bn_relu_f32t<<<CDIV(B * 128 * N1, 256), 256, 0, stream>>>(
      T2, bnScale, bnShift, f1nt, 128, N1, B * 128 * N1);

  // ---- module 0: (xyz0, xyz1, f0, f1nt) -> d_out (B,128,8192) NCHW f32
  interp_concat<<<dim3(N0 / 256, B), 256, N1 * 3 * 4, stream>>>(
      xyz0, xyz1, f0, f1nt, T1, N0, N1, C0, C1);
  conv_bn(wh00, T1, T2, g00, b00, bnbuf, B, 128, C0 + C1, N0, stream);
  bn_relu_f16t<<<CDIV(B * 128 * N0, 256), 256, 0, stream>>>(
      T2, bnScale, bnShift, T1, 128, N0, B * 128 * N0);
  conv_bn(wh01, T1, T2, g01, b01, bnbuf, B, 128, 128, N0, stream);
  bn_relu_nchw<<<CDIV(B * 128 * N0, 256), 256, 0, stream>>>(
      T2, bnScale, bnShift, (float*)d_out, 128, N0, B * 128 * N0);
}
